// RGM_5153960755770
// MI455X (gfx1250) — compile-verified, matrix operand order checked by reading, not run on hardware
//
#include <hip/hip_runtime.h>

#define B_   32
#define T_   4096
#define DIN  256
#define DH   256

typedef __attribute__((ext_vector_type(16))) __bf16         v16bf;
typedef __attribute__((ext_vector_type(8)))  float          v8f;
typedef __attribute__((ext_vector_type(8)))  unsigned int   v8i;
typedef __attribute__((ext_vector_type(4)))  unsigned int   v4i;

// Pack two f32 into one dword of bf16 (lo -> [15:0], hi -> [31:16]).
__device__ __forceinline__ unsigned pk_bf16(float lo, float hi) {
#if __has_builtin(__builtin_amdgcn_cvt_pk_bf16_f32)
    auto r = __builtin_amdgcn_cvt_pk_bf16_f32(lo, hi);   // v_cvt_pk_bf16_f32, RNE
    return __builtin_bit_cast(unsigned, r);
#else
    // v_perm_b32: take high 16 bits of each source (truncating bf16 convert)
    return __builtin_amdgcn_perm(__float_as_uint(hi), __float_as_uint(lo), 0x07060302u);
#endif
}

__device__ __forceinline__ unsigned short f2bf(float f) {
    unsigned u = __float_as_uint(f);
    u += 0x7FFFu + ((u >> 16) & 1u);          // round-to-nearest-even
    return (unsigned short)(u >> 16);
}

// Build a 16-element bf16 fragment from f32 memory.
// Per ISA layout, one lane's fragment = elements q[0..7] and q[16..23].
__device__ __forceinline__ v8i cvt_frag_f32(const float* __restrict__ q) {
    v8i o;
#pragma unroll
    for (int p = 0; p < 4; ++p) {
        o[p]     = pk_bf16(q[2 * p],      q[2 * p + 1]);
        o[4 + p] = pk_bf16(q[16 + 2 * p], q[16 + 2 * p + 1]);
    }
    return o;
}

// Same fragment slice from LDS bf16: two 16B ds_load_b128.
__device__ __forceinline__ v8i load_frag_lds(const unsigned short* q) {
    v4i lo = *(const v4i*)(q);
    v4i hi = *(const v4i*)(q + 16);
    return __builtin_shufflevector(lo, hi, 0, 1, 2, 3, 4, 5, 6, 7);
}

__device__ __forceinline__ v8f wmma_bf16(v8i a, v8i b, v8f c) {
    return __builtin_amdgcn_wmma_f32_16x16x32_bf16(
        false, __builtin_bit_cast(v16bf, a),
        false, __builtin_bit_cast(v16bf, b),
        (short)0, c, false, false);
}

__device__ __forceinline__ float my_tanh(float x) {
#if __has_builtin(__builtin_amdgcn_tanhf)
    return __builtin_amdgcn_tanhf(x);
#else
    return tanhf(x);
#endif
}

// ---------------------------------------------------------------------------
// Phase 1: xproj = input @ Wx^T + b, written into d_out (in place for phase 2)
// Memory-bound (~270 MB @ 23.3 TB/s ~= 12 us). Block = 256 thr, tile 128x64.
// ---------------------------------------------------------------------------
__global__ __launch_bounds__(256, 1) void xproj_kernel(
    const float* __restrict__ input, const float* __restrict__ Wx,
    const float* __restrict__ bias, float* __restrict__ out)
{
    const int lane = threadIdx.x & 31;
    const int wave = threadIdx.x >> 5;          // 0..7
    const int li   = lane & 15;
    const int hi   = lane >> 4;                 // 0/1

    const int m0 = blockIdx.x * 128 + wave * 16;   // row block in B*T space
    const int nb = blockIdx.y * 64;                // 4 N-tiles of 16

    const float* ap = input + (size_t)(m0 + li) * DIN;

    v8f c[4];
#pragma unroll
    for (int nt = 0; nt < 4; ++nt) {
        float bv = bias[nb + nt * 16 + li];
#pragma unroll
        for (int r = 0; r < 8; ++r) c[nt][r] = bv;
    }

#pragma unroll
    for (int k = 0; k < 8; ++k) {
        v8i a = cvt_frag_f32(ap + k * 32 + hi * 8);
#pragma unroll
        for (int nt = 0; nt < 4; ++nt) {
            const float* bp = Wx + (size_t)(nb + nt * 16 + li) * DIN + k * 32 + hi * 8;
            v8i b = cvt_frag_f32(bp);
            c[nt] = wmma_bf16(a, b, c[nt]);
        }
    }

    // C layout: VGPR r -> row m0 + r + hi*8, col nb + nt*16 + li
#pragma unroll
    for (int nt = 0; nt < 4; ++nt) {
        float* op = out + (size_t)(m0 + hi * 8) * DH + nb + nt * 16 + li;
#pragma unroll
        for (int r = 0; r < 8; ++r) op[(size_t)r * DH] = c[nt][r];
    }
}

// ---------------------------------------------------------------------------
// Phase 2: sequential scan z = tanh(xproj_t + z @ Wh^T), latency-bound.
// 2 blocks (16 batch rows each) x 512 thr (16 waves, one 16x16 N-tile each).
// Wh lives in registers (bf16), z ping-pongs in LDS as bf16, one barrier/step.
// Two WMMA accumulator chains halve the serialized D->C dependency depth.
// ---------------------------------------------------------------------------
__global__ __launch_bounds__(512, 1) void rnn_scan_kernel(
    const float* __restrict__ z0, const float* __restrict__ Wh,
    float* __restrict__ out /* xproj in, activations out */,
    float* __restrict__ zn)
{
    __shared__ alignas(16) unsigned short zbuf[2][16][DH + 8]; // +8 ushort pad vs bank conflicts

    const int tid  = threadIdx.x;
    const int lane = tid & 31;
    const int wave = tid >> 5;                  // 0..15
    const int li   = lane & 15;
    const int hi   = lane >> 4;
    const int b0   = blockIdx.x * 16;           // this WG's batch rows
    const int n0   = wave * 16;                 // this wave's N tile

    // Wh B-fragments for this N tile, bf16 in registers, reused all T steps.
    v8i bfrag[8];
#pragma unroll
    for (int k = 0; k < 8; ++k)
        bfrag[k] = cvt_frag_f32(Wh + (size_t)(n0 + li) * DH + k * 32 + hi * 8);

    // Seed zbuf[0] = bf16(z0)
    for (int e = tid; e < 16 * DH; e += 512) {
        int row = e >> 8, col = e & (DH - 1);
        zbuf[0][row][col] = f2bf(z0[(size_t)(b0 + row) * DH + col]);
    }
    __syncthreads();

    // Per-lane base pointer into xproj/out: row b0 + hi*8 + r, col n0 + li
    float* base = out + (size_t)(b0 + hi * 8) * T_ * DH + n0 + li;
    const size_t rstride = (size_t)T_ * DH;

    int cur = 0;
    for (int t = 0; t < T_; ++t) {
        float* pt = base + (size_t)t * DH;

        // C init = xproj tile (issue global loads first: longest latency)
        v8f c;
#pragma unroll
        for (int r = 0; r < 8; ++r) c[r] = pt[(size_t)r * rstride];

        // A fragments: bf16 z from LDS (16 x ds_load_b128)
        v8i a[8];
#pragma unroll
        for (int k = 0; k < 8; ++k)
            a[k] = load_frag_lds(&zbuf[cur][li][k * 32 + hi * 8]);

        // Two independent accumulation chains
        v8f c2 = {};
#pragma unroll
        for (int k = 0; k < 4; ++k) {
            c  = wmma_bf16(a[2 * k],     bfrag[2 * k],     c);
            c2 = wmma_bf16(a[2 * k + 1], bfrag[2 * k + 1], c2);
        }

        // Prefetch next timestep's xproj while we finish this step
        if (t + 1 < T_) __builtin_prefetch(pt + DH, 0, 0);

#pragma unroll
        for (int r = 0; r < 8; ++r) c[r] = my_tanh(c[r] + c2[r]);

        // Store activations (overwrite xproj in place)
#pragma unroll
        for (int r = 0; r < 8; ++r) pt[(size_t)r * rstride] = c[r];

        // Publish bf16 z_new for next step
        const int nxt = cur ^ 1;
#pragma unroll
        for (int r = 0; r < 8; ++r)
            zbuf[nxt][hi * 8 + r][n0 + li] = f2bf(c[r]);

        if (t == T_ - 1) {
#pragma unroll
            for (int r = 0; r < 8; ++r)
                zn[(size_t)(b0 + hi * 8 + r) * DH + n0 + li] = c[r];
        }

        __syncthreads();
        cur = nxt;
    }
}

extern "C" void kernel_launch(void* const* d_in, const int* in_sizes, int n_in,
                              void* d_out, int out_size, void* d_ws, size_t ws_size,
                              hipStream_t stream) {
    (void)in_sizes; (void)n_in; (void)out_size; (void)d_ws; (void)ws_size;
    const float* input = (const float*)d_in[0];
    const float* z0    = (const float*)d_in[1];
    const float* Wx    = (const float*)d_in[2];
    const float* Wh    = (const float*)d_in[3];
    const float* bias  = (const float*)d_in[4];

    float* out = (float*)d_out;
    float* zn  = out + (size_t)B_ * T_ * DH;

    dim3 g1((B_ * T_) / 128, DH / 64);
    xproj_kernel<<<g1, 256, 0, stream>>>(input, Wx, bias, out);
    rnn_scan_kernel<<<dim3(2), 512, 0, stream>>>(z0, Wh, out, zn);
}
